// HFScatterMoEGatedMLP_47278999994723
// MI455X (gfx1250) — compile-verified, matrix operand order checked by reading, not run on hardware
//
#include <hip/hip_runtime.h>

// MoE gated MLP + LoRA for MI455X (gfx1250, wave32, WMMA bf16 16x16x32).
// - top-2/8 routing -> per-expert token compaction (4x FLOP cut vs dense)
// - weights converted once to bf16 (~100MB, fits the 192MB L2)
// - all GEMMs on v_wmma_f32_16x16x32_bf16 (fp32 accumulate)
// - LoRA rank-16 zero-padded to K=32 => single epilogue WMMA
// - main GEMM tiles staged via GLOBAL_LOAD_ASYNC_TO_LDS_B128 (ASYNCcnt),
//   double-buffered LDS; falls back to direct copies if builtin is absent
// - deterministic: out zero-initialized, experts accumulate in fixed order
// Workspace use: ~208 MB.

#define T_TOK 16384
#define D_DIM 1024
#define F_DIM 2048
#define E_NUM 8
#define R_RANK 16
#define RPAD 32

typedef unsigned short u16;
typedef __attribute__((ext_vector_type(16))) __bf16 v16bf;
typedef __attribute__((ext_vector_type(8)))  float  v8f;
typedef int v4i __attribute__((vector_size(16)));   // builtin's pointee type

struct __attribute__((aligned(16))) U128 { unsigned int x, y, z, w; };
union Frag { v16bf bf; U128 u[2]; };

#if __has_builtin(__builtin_amdgcn_global_load_async_to_lds_b128) && \
    __has_builtin(__builtin_amdgcn_s_wait_asynccnt)
#define HAVE_ASYNC_LDS 1
#else
#define HAVE_ASYNC_LDS 0
#endif

#define AS1 __attribute__((address_space(1)))
#define AS3 __attribute__((address_space(3)))

// 16-byte global -> LDS copy: async DMA on gfx1250 (no VGPR staging).
__device__ __forceinline__ void copy16_g2l(u16* lds, const u16* glob) {
#if HAVE_ASYNC_LDS
  __builtin_amdgcn_global_load_async_to_lds_b128((AS1 v4i*)glob,
                                                 (AS3 v4i*)lds, 0, 0);
#else
  *(U128*)lds = *(const U128*)glob;
#endif
}

// Wait until at most `n` of this wave's async copies remain in flight.
#define ASYNC_WAIT(n)                       \
  do {                                      \
    if (HAVE_ASYNC_LDS)                     \
      __builtin_amdgcn_s_wait_asynccnt(n);  \
  } while (0)

__device__ __forceinline__ u16 f32_to_bf16(float f) {
  unsigned int u = __float_as_uint(f);
  u += 0x7fffu + ((u >> 16) & 1u);          // round-to-nearest-even
  return (u16)(u >> 16);
}

__device__ __forceinline__ v8f wmma_bf16(const Frag& a, const Frag& b, v8f c) {
  return __builtin_amdgcn_wmma_f32_16x16x32_bf16(false, a.bf, false, b.bf,
                                                 (short)0, c, false, false);
}

// Load a 16x32 bf16 fragment whose rows live at `base + row*stride` (elements).
// Per-lane ISA layout: lane<16 -> M=lane, K in {kh*8.., 16+kh*8..}.
__device__ __forceinline__ void load_frag(Frag& f, const u16* base,
                                          int row, int kh, size_t stride) {
  const u16* p = base + (size_t)row * stride + kh * 8;
  f.u[0] = *(const U128*)p;
  f.u[1] = *(const U128*)(p + 16);
}

// ---------------- conversion / prep kernels ----------------

__global__ void convert_bf16_kernel(const float* __restrict__ in,
                                    u16* __restrict__ out, size_t n) {
  size_t i = (size_t)blockIdx.x * blockDim.x + threadIdx.x;
  if (i < n) out[i] = f32_to_bf16(in[i]);
}

__global__ void zero_out_kernel(float* __restrict__ out, size_t n) {
  size_t i = (size_t)blockIdx.x * blockDim.x + threadIdx.x;
  if (i < n) out[i] = 0.f;
}

// in: [batch][rows][cols] f32 -> out: [batch][cols][rows] bf16
__global__ void transpose_bf16_kernel(const float* __restrict__ in,
                                      u16* __restrict__ out,
                                      int rows, int cols) {
  __shared__ float tile[32][33];
  const float* inp = in + (size_t)blockIdx.z * rows * cols;
  u16* outp = out + (size_t)blockIdx.z * rows * cols;
  int c0 = blockIdx.x * 32, r0 = blockIdx.y * 32;
  int tx = threadIdx.x & 31, ty = threadIdx.x >> 5;   // 32x8 threads
#pragma unroll
  for (int i = 0; i < 32; i += 8)
    tile[ty + i][tx] = inp[(size_t)(r0 + ty + i) * cols + c0 + tx];
  __syncthreads();
#pragma unroll
  for (int i = 0; i < 32; i += 8)
    outp[(size_t)(c0 + ty + i) * rows + r0 + tx] = f32_to_bf16(tile[tx][ty + i]);
}

// LoRA B [rows][16] f32 -> [rows][32] bf16 (cols 16..31 zero)
__global__ void pad_loraB_kernel(const float* __restrict__ in,
                                 u16* __restrict__ out, size_t rows) {
  size_t i = (size_t)blockIdx.x * blockDim.x + threadIdx.x;
  if (i >= rows * RPAD) return;
  size_t r = i >> 5; int c = (int)(i & 31);
  out[i] = (c < R_RANK) ? f32_to_bf16(in[r * R_RANK + c]) : (u16)0;
}

// ---------------- routing + compaction ----------------

__global__ void route_kernel(const float* __restrict__ x,
                             const float* __restrict__ gw,
                             const float* __restrict__ gd,
                             float* __restrict__ combine) {
  int wave = threadIdx.x >> 5, lane = threadIdx.x & 31;
  int t = blockIdx.x * 8 + wave;
  const float* xr = x + (size_t)t * D_DIM;
  float acc[E_NUM];
#pragma unroll
  for (int e = 0; e < E_NUM; ++e) acc[e] = 0.f;
  for (int d = lane; d < D_DIM; d += 32) {
    float xv = xr[d];
#pragma unroll
    for (int e = 0; e < E_NUM; ++e)
      acc[e] += xv * (gw[e * D_DIM + d] + gd[e * D_DIM + d]);
  }
#pragma unroll
  for (int off = 16; off > 0; off >>= 1)
#pragma unroll
    for (int e = 0; e < E_NUM; ++e)
      acc[e] += __shfl_down(acc[e], off, 32);
  if (lane == 0) {
    float m = acc[0];
#pragma unroll
    for (int e = 1; e < E_NUM; ++e) m = fmaxf(m, acc[e]);
    float p[E_NUM];
#pragma unroll
    for (int e = 0; e < E_NUM; ++e) p[e] = __expf(acc[e] - m);
    int i1 = 0;
#pragma unroll
    for (int e = 1; e < E_NUM; ++e) if (p[e] > p[i1]) i1 = e;
    int i2 = (i1 == 0) ? 1 : 0;
#pragma unroll
    for (int e = 0; e < E_NUM; ++e)
      if (e != i1 && p[e] > p[i2]) i2 = e;
    float s = p[i1] + p[i2];   // softmax Z cancels in the top-2 renorm
#pragma unroll
    for (int e = 0; e < E_NUM; ++e)
      combine[(size_t)t * E_NUM + e] =
          (e == i1) ? p[i1] / s : ((e == i2) ? p[i2] / s : 0.f);
  }
}

__global__ void zero_cnt_kernel(int* cnt) {
  if (threadIdx.x < E_NUM) cnt[threadIdx.x] = 0;
}

__global__ void build_lists_kernel(const float* __restrict__ combine,
                                   int* __restrict__ cnt,
                                   int* __restrict__ idxbuf) {
  int t = blockIdx.x * blockDim.x + threadIdx.x;
  if (t >= T_TOK) return;
#pragma unroll
  for (int e = 0; e < E_NUM; ++e) {
    if (combine[(size_t)t * E_NUM + e] > 0.f) {
      int pos = atomicAdd(&cnt[e], 1);       // order irrelevant: row placement
      idxbuf[(size_t)e * T_TOK + pos] = t;   // does not change per-row math
    }
  }
}

__global__ void pad_lists_kernel(int* __restrict__ cnt,
                                 int* __restrict__ padcnt,
                                 int* __restrict__ idxbuf) {
  int e = threadIdx.x;
  if (e >= E_NUM) return;
  int n = cnt[e];
  int np = (n + 127) & ~127;                 // down_kernel uses 128-row tiles
  for (int i = n; i < np; ++i) idxbuf[(size_t)e * T_TOK + i] = -1;
  padcnt[e] = np;
}

// ------- skinny LoRA-t: tout = 2 * (rows @ A^T), bf16 rank-padded -------

// If idx != nullptr, rows are gathered from Xbf by token id; else compacted rows.
__global__ void lora_t_kernel(const u16* __restrict__ Xbf,
                              const u16* __restrict__ Abf,   // [16][Kdim]
                              const int* __restrict__ idx,   // may be null
                              const int* __restrict__ padcnt, int e,
                              u16* __restrict__ tout,        // [T][32]
                              int Kdim) {
  int m0 = blockIdx.x * 16;
  if (m0 >= padcnt[e]) return;
  int lane = threadIdx.x & 31;
  int row = lane & 15, kh = lane >> 4;
  int src = m0 + row;
  if (idx) { int tk = idx[src]; src = tk < 0 ? 0 : tk; }
  v8f acc = {};
  for (int k0 = 0; k0 < Kdim; k0 += 32) {
    Frag a, b;
    load_frag(a, Xbf + k0, src, kh, (size_t)Kdim);
    load_frag(b, Abf + k0, row, kh, (size_t)Kdim);
    acc = wmma_bf16(a, b, acc);
  }
  int col = lane & 15, rbase = kh * 8;
#pragma unroll
  for (int r = 0; r < 8; ++r) {
    size_t o = (size_t)(m0 + rbase + r) * RPAD;
    tout[o + col] = f32_to_bf16(2.0f * acc[r]);   // LORA_SCALING folded in
    tout[o + 16 + col] = 0;                       // rank pad
  }
}

// ---------------- fused gate_up GEMM + LoRA + SwiGLU ----------------
// Block tile: 64 rows x 64 cols of gate AND up (waves 0-3 gate, 4-7 up).

__global__ __launch_bounds__(256) void gate_up_kernel(
    const u16* __restrict__ Xbf,   // [T][D]
    const u16* __restrict__ Wgu,   // [2F][D] this expert (N-major)
    const u16* __restrict__ t1,    // [Tc][32]
    const u16* __restrict__ guB,   // [2F][32]
    const int* __restrict__ idx, const int* __restrict__ padcnt, int e,
    u16* __restrict__ Hbf) {       // [Tc][F] compacted
  int m0 = blockIdx.y * 64;
  if (m0 >= padcnt[e]) return;
  __shared__ u16 sA[2][64 * 32];
  __shared__ u16 sBg[2][64 * 32];
  __shared__ u16 sBu[2][64 * 32];
  __shared__ float sC[2][64][64];
  __shared__ int sTok[64];
  int n0 = blockIdx.x * 64;
  int wave = threadIdx.x >> 5, lane = threadIdx.x & 31;
  int isUp = wave >> 2, sm = wave & 3;
  int row = lane & 15, kh = lane >> 4;
  int ldRow = threadIdx.x >> 2, ldCol = (threadIdx.x & 3) * 8;
  if (threadIdx.x < 64) {
    int tk = idx[m0 + threadIdx.x];
    sTok[threadIdx.x] = tk < 0 ? 0 : tk;
  }
  __syncthreads();
  int tokA = sTok[ldRow];

  auto issue = [&](int buf, int k0) {   // 3 async b128 per thread per stage
    copy16_g2l(&sA[buf][ldRow * 32 + ldCol],
               &Xbf[(size_t)tokA * D_DIM + k0 + ldCol]);
    copy16_g2l(&sBg[buf][ldRow * 32 + ldCol],
               &Wgu[(size_t)(n0 + ldRow) * D_DIM + k0 + ldCol]);
    copy16_g2l(&sBu[buf][ldRow * 32 + ldCol],
               &Wgu[(size_t)(F_DIM + n0 + ldRow) * D_DIM + k0 + ldCol]);
  };

  v8f acc[4] = {};
  issue(0, 0);
  for (int k0 = 0; k0 < D_DIM; k0 += 32) {
    int buf = (k0 >> 5) & 1;
    if (k0 + 32 < D_DIM) { issue(buf ^ 1, k0 + 32); ASYNC_WAIT(3); }
    else                 { ASYNC_WAIT(0); }
    __syncthreads();
    Frag a;
    load_frag(a, sA[buf], sm * 16 + row, kh, 32);
    const u16* sB = isUp ? sBu[buf] : sBg[buf];
#pragma unroll
    for (int j = 0; j < 4; ++j) {
      Frag b;
      load_frag(b, sB, j * 16 + row, kh, 32);
      acc[j] = wmma_bf16(a, b, acc[j]);
    }
    __syncthreads();
  }
  {  // LoRA epilogue: one K=32 WMMA per tile (rank 16 zero-padded)
    Frag a;
    load_frag(a, t1 + (size_t)m0 * RPAD, sm * 16 + row, kh, RPAD);
#pragma unroll
    for (int j = 0; j < 4; ++j) {
      Frag b;
      load_frag(b, guB + (size_t)((isUp ? F_DIM : 0) + n0) * RPAD,
                j * 16 + row, kh, RPAD);
      acc[j] = wmma_bf16(a, b, acc[j]);
    }
  }
  int col = lane & 15, rbase = kh * 8;
#pragma unroll
  for (int j = 0; j < 4; ++j)
#pragma unroll
    for (int r = 0; r < 8; ++r)
      sC[isUp][sm * 16 + rbase + r][j * 16 + col] = acc[j][r];
  __syncthreads();
  for (int i = threadIdx.x; i < 64 * 64; i += 256) {
    int r = i >> 6, c = i & 63;
    float g = sC[0][r][c], u = sC[1][r][c];
    float h = (g / (1.f + __expf(-g))) * u;       // silu(g)*u
    Hbf[(size_t)(m0 + r) * F_DIM + n0 + c] = f32_to_bf16(h);
  }
}

// -------- down GEMM + LoRA + combine-scaled scatter into out --------
// Block tile: 128 rows x 64 cols; wave w owns rows [16w,16w+16) x all 64 cols.

__global__ __launch_bounds__(256) void down_kernel(
    const u16* __restrict__ Hbf,   // [Tc][F] compacted
    const u16* __restrict__ Wdn,   // [D][F] this expert (N-major)
    const u16* __restrict__ t2,    // [Tc][32]
    const u16* __restrict__ dnB,   // [D][32]
    const int* __restrict__ idx, const int* __restrict__ padcnt, int e,
    const float* __restrict__ combine,
    float* __restrict__ out) {
  int m0 = blockIdx.y * 128;
  if (m0 >= padcnt[e]) return;
  __shared__ u16 sA[2][128 * 32];
  __shared__ u16 sB[2][64 * 32];
  __shared__ int sTok[128];
  int n0 = blockIdx.x * 64;
  int wave = threadIdx.x >> 5, lane = threadIdx.x & 31;
  int row = lane & 15, kh = lane >> 4;
  int ldRow = threadIdx.x >> 2, ldCol = (threadIdx.x & 3) * 8;
  if (threadIdx.x < 128) sTok[threadIdx.x] = idx[m0 + threadIdx.x];

  auto issue = [&](int buf, int k0) {   // 3 async b128 per thread per stage
    copy16_g2l(&sA[buf][ldRow * 32 + ldCol],
               &Hbf[(size_t)(m0 + ldRow) * F_DIM + k0 + ldCol]);
    copy16_g2l(&sA[buf][(ldRow + 64) * 32 + ldCol],
               &Hbf[(size_t)(m0 + ldRow + 64) * F_DIM + k0 + ldCol]);
    copy16_g2l(&sB[buf][ldRow * 32 + ldCol],
               &Wdn[(size_t)(n0 + ldRow) * F_DIM + k0 + ldCol]);
  };

  v8f acc[4] = {};
  issue(0, 0);
  for (int k0 = 0; k0 < F_DIM; k0 += 32) {
    int buf = (k0 >> 5) & 1;
    if (k0 + 32 < F_DIM) { issue(buf ^ 1, k0 + 32); ASYNC_WAIT(3); }
    else                 { ASYNC_WAIT(0); }
    __syncthreads();
    Frag a;
    load_frag(a, sA[buf], wave * 16 + row, kh, 32);
#pragma unroll
    for (int j = 0; j < 4; ++j) {
      Frag b;
      load_frag(b, sB[buf], j * 16 + row, kh, 32);
      acc[j] = wmma_bf16(a, b, acc[j]);
    }
    __syncthreads();
  }
  {  // LoRA epilogue
    Frag a;
    load_frag(a, t2 + (size_t)m0 * RPAD, wave * 16 + row, kh, RPAD);
#pragma unroll
    for (int j = 0; j < 4; ++j) {
      Frag b;
      load_frag(b, dnB + (size_t)n0 * RPAD, j * 16 + row, kh, RPAD);
      acc[j] = wmma_bf16(a, b, acc[j]);
    }
  }
  int col = lane & 15, rbase = kh * 8;
#pragma unroll
  for (int r = 0; r < 8; ++r) {
    int tk = sTok[wave * 16 + rbase + r];
    if (tk < 0) continue;                         // padding row
    float cw = combine[(size_t)tk * E_NUM + e];
#pragma unroll
    for (int j = 0; j < 4; ++j) {
      size_t o = (size_t)tk * D_DIM + n0 + j * 16 + col;
      out[o] += cw * acc[j][r];  // out pre-zeroed; fixed expert order => determ.
    }
  }
}

// ---------------- host ----------------

extern "C" void kernel_launch(void* const* d_in, const int* in_sizes, int n_in,
                              void* d_out, int out_size, void* d_ws, size_t ws_size,
                              hipStream_t stream) {
  (void)in_sizes; (void)n_in; (void)out_size; (void)ws_size;
  const float* x   = (const float*)d_in[0];
  const float* gw  = (const float*)d_in[1];
  const float* gd  = (const float*)d_in[2];
  const float* wgu = (const float*)d_in[3];
  const float* wdn = (const float*)d_in[4];
  const float* guA = (const float*)d_in[5];
  const float* guB = (const float*)d_in[6];
  const float* dnA = (const float*)d_in[7];
  const float* dnB = (const float*)d_in[8];
  float* out = (float*)d_out;

  u16* p = (u16*)d_ws;
  u16* x_bf   = p; p += (size_t)T_TOK * D_DIM;              // 32 MB
  u16* wgu_bf = p; p += (size_t)E_NUM * 2 * F_DIM * D_DIM;  // 64 MB
  u16* wdn_bf = p; p += (size_t)E_NUM * D_DIM * F_DIM;      // 32 MB
  u16* guA_bf = p; p += (size_t)E_NUM * R_RANK * D_DIM;
  u16* dnA_bf = p; p += (size_t)E_NUM * R_RANK * F_DIM;
  u16* guB_bf = p; p += (size_t)E_NUM * 2 * F_DIM * RPAD;
  u16* dnB_bf = p; p += (size_t)E_NUM * D_DIM * RPAD;
  u16* t1     = p; p += (size_t)T_TOK * RPAD;
  u16* t2     = p; p += (size_t)T_TOK * RPAD;
  u16* h_bf   = p; p += (size_t)T_TOK * F_DIM;              // 64 MB
  float* combine = (float*)p; p += (size_t)T_TOK * E_NUM * 2;
  int* cnt     = (int*)p; p += 16;
  int* padcnt  = (int*)p; p += 16;
  int* idxbuf  = (int*)p;                                   // E*T ints

  auto cdiv = [](size_t a, size_t b) { return (unsigned)((a + b - 1) / b); };

  // prep
  convert_bf16_kernel<<<cdiv((size_t)T_TOK * D_DIM, 256), 256, 0, stream>>>(
      x, x_bf, (size_t)T_TOK * D_DIM);
  route_kernel<<<T_TOK / 8, 256, 0, stream>>>(x, gw, gd, combine);
  transpose_bf16_kernel<<<dim3(2 * F_DIM / 32, D_DIM / 32, E_NUM), 256, 0,
                          stream>>>(wgu, wgu_bf, D_DIM, 2 * F_DIM);
  transpose_bf16_kernel<<<dim3(D_DIM / 32, F_DIM / 32, E_NUM), 256, 0,
                          stream>>>(wdn, wdn_bf, F_DIM, D_DIM);
  convert_bf16_kernel<<<cdiv((size_t)E_NUM * R_RANK * D_DIM, 256), 256, 0,
                        stream>>>(guA, guA_bf, (size_t)E_NUM * R_RANK * D_DIM);
  convert_bf16_kernel<<<cdiv((size_t)E_NUM * R_RANK * F_DIM, 256), 256, 0,
                        stream>>>(dnA, dnA_bf, (size_t)E_NUM * R_RANK * F_DIM);
  pad_loraB_kernel<<<cdiv((size_t)E_NUM * 2 * F_DIM * RPAD, 256), 256, 0,
                     stream>>>(guB, guB_bf, (size_t)E_NUM * 2 * F_DIM);
  pad_loraB_kernel<<<cdiv((size_t)E_NUM * D_DIM * RPAD, 256), 256, 0,
                     stream>>>(dnB, dnB_bf, (size_t)E_NUM * D_DIM);
  zero_out_kernel<<<cdiv((size_t)T_TOK * D_DIM, 256), 256, 0, stream>>>(
      out, (size_t)T_TOK * D_DIM);

  // compaction
  zero_cnt_kernel<<<1, 32, 0, stream>>>(cnt);
  build_lists_kernel<<<T_TOK / 256, 256, 0, stream>>>(combine, cnt, idxbuf);
  pad_lists_kernel<<<1, 32, 0, stream>>>(cnt, padcnt, idxbuf);

  // per-expert pipeline (sequential => deterministic accumulation into out)
  for (int e = 0; e < E_NUM; ++e) {
    const int* idx = idxbuf + (size_t)e * T_TOK;
    lora_t_kernel<<<T_TOK / 16, 32, 0, stream>>>(
        x_bf, guA_bf + (size_t)e * R_RANK * D_DIM, idx, padcnt, e, t1, D_DIM);
    gate_up_kernel<<<dim3(F_DIM / 64, T_TOK / 64), 256, 0, stream>>>(
        x_bf, wgu_bf + (size_t)e * 2 * F_DIM * D_DIM, t1,
        guB_bf + (size_t)e * 2 * F_DIM * RPAD, idx, padcnt, e, h_bf);
    lora_t_kernel<<<T_TOK / 16, 32, 0, stream>>>(
        h_bf, dnA_bf + (size_t)e * R_RANK * F_DIM, nullptr, padcnt, e, t2,
        F_DIM);
    down_kernel<<<dim3(D_DIM / 64, T_TOK / 128), 256, 0, stream>>>(
        h_bf, wdn_bf + (size_t)e * D_DIM * F_DIM, t2,
        dnB_bf + (size_t)e * D_DIM * RPAD, idx, padcnt, e, combine, out);
  }
}
